// MambaBottleneck_66305705115707
// MI455X (gfx1250) — matrix operand order repaired and checked by reading, not compiled, not run on hardware
//
#include <hip/hip_runtime.h>
#include <hip/hip_bf16.h>

// ---------------------------------------------------------------------------
// Mamba block (B=8, L=4096, d_model=256, d_inner=512, d_state=16, dt_rank=16)
// for MI455X / gfx1250 (wave32, WMMA bf16 16x16x32, f32 accumulate).
//
// Pipeline (all on `stream`):
//   0) prep: weights f32 -> bf16, A = -exp(A_log)
//   1) in_proj GEMM (WMMA)          -> u, z
//   2) causal depthwise conv4+SiLU  -> u_c
//   3) x_proj GEMM (WMMA, N=48)     -> x_dbl (dt | B | C)
//   4) dt_proj (K=16, VALU) + softplus -> delta
//   5) chunked selective scan, 3 passes (chunk=128, 32 chunks):
//        pass1: per-chunk decay product P and local end-state
//        pass2: sequential stitch over 32 chunks -> per-chunk init states
//        pass3: replay with init states, fuse y = <h,C> + D*u, gate silu(z)
//   6) out_proj GEMM (WMMA) + residual -> h_pre
//   7) LayerNorm (wave32 shuffle reduction) -> d_out (f32)
// ---------------------------------------------------------------------------

#define BATCH   8
#define SEQLEN  4096
#define DMODEL  256
#define DINNER  512
#define DSTATE  16
#define DTRANK  16
#define MROWS   (BATCH * SEQLEN)      // 32768
#define XDBL_N  (DTRANK + 2 * DSTATE) // 48
#define CHUNK   128
#define NCH     (SEQLEN / CHUNK)      // 32

typedef __attribute__((ext_vector_type(16))) __bf16 v16bf;
typedef __attribute__((ext_vector_type(8)))  float  v8f;
typedef __attribute__((ext_vector_type(8)))  unsigned int v8u;

static __device__ __forceinline__ __bf16 f2bf(float f) {
  unsigned u = __builtin_bit_cast(unsigned, f);
  u += 0x7FFFu + ((u >> 16) & 1u);            // round-to-nearest-even
  unsigned short h = (unsigned short)(u >> 16);
  return __builtin_bit_cast(__bf16, h);
}

static __device__ __forceinline__ float silu_f(float x) {
  return x / (1.0f + __expf(-x));
}

// ---- WMMA fragment loaders (contiguous k order, same for A and B, wave32) --------------------
// A (16x32 bf16, from f32 row-major X[lda]): lane row = m0 + lane%16,
//   kbase = k0 + 16*(lane>=16); elems 0..15 -> K kbase..kbase+15.
static __device__ __forceinline__ v16bf load_a_f32(const float* X, int m0,
                                                   int k0, int lda) {
  const int lane = threadIdx.x & 31;
  const int row  = m0 + (lane & 15);
  const int kb   = k0 + ((lane >> 4) << 4);
  const float* p = X + (size_t)row * lda + kb;
  float4 f0 = *(const float4*)(p);
  float4 f1 = *(const float4*)(p + 4);
  float4 f2 = *(const float4*)(p + 8);
  float4 f3 = *(const float4*)(p + 12);
  v16bf a;
  a[0]=f2bf(f0.x); a[1]=f2bf(f0.y); a[2]=f2bf(f0.z); a[3]=f2bf(f0.w);
  a[4]=f2bf(f1.x); a[5]=f2bf(f1.y); a[6]=f2bf(f1.z); a[7]=f2bf(f1.w);
  a[8]=f2bf(f2.x); a[9]=f2bf(f2.y); a[10]=f2bf(f2.z); a[11]=f2bf(f2.w);
  a[12]=f2bf(f3.x); a[13]=f2bf(f3.y); a[14]=f2bf(f3.z); a[15]=f2bf(f3.w);
  return a;
}

// B (32x16 bf16, from bf16 row-major W[N][ldb], column n = output feature):
//   lane col = n0 + lane%16, kbase = k0 + 16*(lane>=16), elems = K kbase..+15.
static __device__ __forceinline__ v16bf load_b_bf16(const __bf16* W, int n0,
                                                    int k0, int ldb) {
  const int lane = threadIdx.x & 31;
  const int n  = n0 + (lane & 15);
  const int kb = k0 + ((lane >> 4) << 4);
  v8u raw = *(const v8u*)(W + (size_t)n * ldb + kb);   // 32B contiguous
  return __builtin_bit_cast(v16bf, raw);
}

// ---- 0) weight prep --------------------------------------------------------
#define NIN  (1024 * 256)
#define NXP  (48 * 512)
#define NOUT (256 * 512)
#define NA   (512 * 16)

__global__ __launch_bounds__(256)
void prep_kernel(const float* w_in, const float* w_xp, const float* w_out,
                 const float* a_log, __bf16* wb_in, __bf16* wb_xp,
                 __bf16* wb_out, float* a_neg) {
  int i = blockIdx.x * blockDim.x + threadIdx.x;
  int stride = gridDim.x * blockDim.x;
  for (; i < NIN + NXP + NOUT + NA; i += stride) {
    if (i < NIN)                   wb_in[i] = f2bf(w_in[i]);
    else if (i < NIN + NXP)        wb_xp[i - NIN] = f2bf(w_xp[i - NIN]);
    else if (i < NIN + NXP + NOUT) wb_out[i - NIN - NXP] = f2bf(w_out[i - NIN - NXP]);
    else { int j = i - NIN - NXP - NOUT; a_neg[j] = -__expf(a_log[j]); }
  }
}

// ---- 1) in_proj GEMM: xz = x(M x 256) @ W^T(256 x 1024) -> u | z -----------
// block 128 = 4 waves; wave computes a 16(M) x 64(N) strip (4 wmma acc).
__global__ __launch_bounds__(128)
void inproj_gemm_kernel(const float* X, const __bf16* Wb, float* U, float* Z) {
  const int wave = threadIdx.x >> 5;
  const int m0 = blockIdx.x * 16;
  const int n0 = (blockIdx.y * 4 + wave) * 64;
  v8f acc[4] = {};
  for (int k0 = 0; k0 < DMODEL; k0 += 32) {
    v16bf a = load_a_f32(X, m0, k0, DMODEL);
#pragma unroll
    for (int j = 0; j < 4; ++j) {
      v16bf b = load_b_bf16(Wb, n0 + j * 16, k0, DMODEL);
      acc[j] = __builtin_amdgcn_wmma_f32_16x16x32_bf16(
          false, a, false, b, (short)0, acc[j], false, false);
    }
  }
  const int lane = threadIdx.x & 31;
  const int rbase = m0 + ((lane >> 4) << 3);
#pragma unroll
  for (int j = 0; j < 4; ++j) {
    const int tilecol = n0 + j * 16;            // 16-aligned, uniform per wave
    const int col = tilecol + (lane & 15);
    if (tilecol < DINNER) {                     // scalar branch: U vs Z
      float* dst = U + (size_t)rbase * DINNER + col;
#pragma unroll
      for (int v = 0; v < 8; ++v) dst[(size_t)v * DINNER] = acc[j][v];
    } else {
      float* dst = Z + (size_t)rbase * DINNER + (col - DINNER);
#pragma unroll
      for (int v = 0; v < 8; ++v) dst[(size_t)v * DINNER] = acc[j][v];
    }
  }
}

// ---- 2) causal depthwise conv1d (width 4) + SiLU ---------------------------
__global__ __launch_bounds__(256)
void conv_silu_kernel(const float* U, const float* cw, const float* cb,
                      float* UC) {
  int idx = blockIdx.x * blockDim.x + threadIdx.x;
  if (idx >= MROWS * DINNER) return;
  int d = idx & (DINNER - 1);
  int t = (idx / DINNER) & (SEQLEN - 1);
  int b = idx / (DINNER * SEQLEN);
  const float* ub = U + (size_t)b * SEQLEN * DINNER + d;
  float acc = cb[d];
#pragma unroll
  for (int j = 0; j < 4; ++j) {
    int tt = t - 3 + j;
    if (tt >= 0) acc = fmaf(cw[d * 4 + j], ub[(size_t)tt * DINNER], acc);
  }
  UC[idx] = silu_f(acc);
}

// ---- 3) x_proj GEMM: x_dbl = u_c(M x 512) @ W^T(512 x 48) ------------------
// block 256 = 8 waves; wave -> 16(M) x 48(N).
__global__ __launch_bounds__(256)
void xproj_gemm_kernel(const float* UC, const __bf16* Wb, float* XD) {
  const int wave = threadIdx.x >> 5;
  const int m0 = (blockIdx.x * 8 + wave) * 16;
  v8f acc[3] = {};
  for (int k0 = 0; k0 < DINNER; k0 += 32) {
    v16bf a = load_a_f32(UC, m0, k0, DINNER);
#pragma unroll
    for (int j = 0; j < 3; ++j) {
      v16bf b = load_b_bf16(Wb, j * 16, k0, DINNER);
      acc[j] = __builtin_amdgcn_wmma_f32_16x16x32_bf16(
          false, a, false, b, (short)0, acc[j], false, false);
    }
  }
  const int lane = threadIdx.x & 31;
  const int rbase = m0 + ((lane >> 4) << 3);
#pragma unroll
  for (int j = 0; j < 3; ++j) {
    int col = j * 16 + (lane & 15);
#pragma unroll
    for (int v = 0; v < 8; ++v)
      XD[(size_t)(rbase + v) * XDBL_N + col] = acc[j][v];
  }
}

// ---- 4) dt_proj (K=16) + softplus -> delta ---------------------------------
__global__ __launch_bounds__(256)
void dtproj_kernel(const float* XD, const float* dtw, const float* dtb,
                   float* DELTA) {
  int idx = blockIdx.x * blockDim.x + threadIdx.x;
  if (idx >= MROWS * DINNER) return;
  int d = idx & (DINNER - 1);
  int row = idx / DINNER;
  const float* dt = XD + (size_t)row * XDBL_N;   // cols 0..15, broadcast load
  float acc = dtb[d];
#pragma unroll
  for (int r = 0; r < DTRANK; ++r)
    acc = fmaf(dt[r], dtw[d * DTRANK + r], acc);
  DELTA[idx] = (acc > 20.0f) ? acc : log1pf(__expf(acc));
}

// ---- 5) chunked selective scan ---------------------------------------------
// chunk-state layout: [b][chunk][s][d]  (d innermost -> coalesced)
#define CS_IDX(b, c, s, d) ((((size_t)(b) * NCH + (c)) * DSTATE + (s)) * DINNER + (d))

__global__ __launch_bounds__(256)
void scan_pass1_kernel(const float* DELTA, const float* UC, const float* XD,
                       const float* Aneg, float* Pbuf, float* Hend) {
  const int d = blockIdx.x * 256 + threadIdx.x;   // gridDim.x = 2
  const int c = blockIdx.y;                       // chunk
  const int b = blockIdx.z;                       // batch
  float As[DSTATE], h[DSTATE], P[DSTATE];
  const float4* ap = (const float4*)(Aneg + d * DSTATE);
#pragma unroll
  for (int q = 0; q < 4; ++q) {
    float4 f = ap[q];
    As[q*4+0]=f.x; As[q*4+1]=f.y; As[q*4+2]=f.z; As[q*4+3]=f.w;
  }
#pragma unroll
  for (int s = 0; s < DSTATE; ++s) { h[s] = 0.0f; P[s] = 1.0f; }
  const int t0 = c * CHUNK;
  for (int t = t0; t < t0 + CHUNK; ++t) {
    size_t row = (size_t)b * SEQLEN + t;
    if (t + 1 < t0 + CHUNK) {                    // prefetch next timestep
      __builtin_prefetch(XD + (row + 1) * XDBL_N + DTRANK, 0, 3);
      __builtin_prefetch(DELTA + (row + 1) * DINNER + d, 0, 3);
      __builtin_prefetch(UC + (row + 1) * DINNER + d, 0, 3);
    }
    float dlt = DELTA[row * DINNER + d];
    float uv  = UC[row * DINNER + d];
    float du  = dlt * uv;
    const float4* Bp = (const float4*)(XD + row * XDBL_N + DTRANK);
    float Bv[DSTATE];
#pragma unroll
    for (int q = 0; q < 4; ++q) {
      float4 f = Bp[q];
      Bv[q*4+0]=f.x; Bv[q*4+1]=f.y; Bv[q*4+2]=f.z; Bv[q*4+3]=f.w;
    }
#pragma unroll
    for (int s = 0; s < DSTATE; ++s) {
      float r = __expf(dlt * As[s]);
      P[s] *= r;
      h[s] = fmaf(h[s], r, du * Bv[s]);
    }
  }
#pragma unroll
  for (int s = 0; s < DSTATE; ++s) {
    Pbuf[CS_IDX(b, c, s, d)] = P[s];
    Hend[CS_IDX(b, c, s, d)] = h[s];
  }
}

__global__ __launch_bounds__(256)
void scan_pass2_kernel(const float* Pbuf, const float* Hend, float* Hinit) {
  int idx = blockIdx.x * blockDim.x + threadIdx.x;   // B*DINNER = 4096
  if (idx >= BATCH * DINNER) return;
  int d = idx & (DINNER - 1);
  int b = idx / DINNER;
  float h[DSTATE];
#pragma unroll
  for (int s = 0; s < DSTATE; ++s) h[s] = 0.0f;
  for (int c = 0; c < NCH; ++c) {
#pragma unroll
    for (int s = 0; s < DSTATE; ++s) {
      Hinit[CS_IDX(b, c, s, d)] = h[s];
      h[s] = fmaf(Pbuf[CS_IDX(b, c, s, d)], h[s], Hend[CS_IDX(b, c, s, d)]);
    }
  }
}

__global__ __launch_bounds__(256)
void scan_pass3_kernel(const float* DELTA, const float* UC, const float* XD,
                       const float* Aneg, const float* Hinit, const float* Z,
                       const float* Dskip, float* YG) {
  const int d = blockIdx.x * 256 + threadIdx.x;
  const int c = blockIdx.y;
  const int b = blockIdx.z;
  float As[DSTATE], h[DSTATE];
  const float4* ap = (const float4*)(Aneg + d * DSTATE);
#pragma unroll
  for (int q = 0; q < 4; ++q) {
    float4 f = ap[q];
    As[q*4+0]=f.x; As[q*4+1]=f.y; As[q*4+2]=f.z; As[q*4+3]=f.w;
  }
#pragma unroll
  for (int s = 0; s < DSTATE; ++s) h[s] = Hinit[CS_IDX(b, c, s, d)];
  const float dsk = Dskip[d];
  const int t0 = c * CHUNK;
  for (int t = t0; t < t0 + CHUNK; ++t) {
    size_t row = (size_t)b * SEQLEN + t;
    if (t + 1 < t0 + CHUNK) {                    // prefetch next timestep
      __builtin_prefetch(XD + (row + 1) * XDBL_N + DTRANK, 0, 3);
      __builtin_prefetch(DELTA + (row + 1) * DINNER + d, 0, 3);
      __builtin_prefetch(UC + (row + 1) * DINNER + d, 0, 3);
      __builtin_prefetch(Z + (row + 1) * DINNER + d, 0, 3);
    }
    float dlt = DELTA[row * DINNER + d];
    float uv  = UC[row * DINNER + d];
    float du  = dlt * uv;
    const float4* Bp = (const float4*)(XD + row * XDBL_N + DTRANK);
    const float4* Cp = (const float4*)(XD + row * XDBL_N + DTRANK + DSTATE);
    float Bv[DSTATE], Cv[DSTATE];
#pragma unroll
    for (int q = 0; q < 4; ++q) {
      float4 f = Bp[q];
      Bv[q*4+0]=f.x; Bv[q*4+1]=f.y; Bv[q*4+2]=f.z; Bv[q*4+3]=f.w;
      float4 g = Cp[q];
      Cv[q*4+0]=g.x; Cv[q*4+1]=g.y; Cv[q*4+2]=g.z; Cv[q*4+3]=g.w;
    }
    float y = 0.0f;
#pragma unroll
    for (int s = 0; s < DSTATE; ++s) {
      float r = __expf(dlt * As[s]);
      h[s] = fmaf(h[s], r, du * Bv[s]);
      y = fmaf(h[s], Cv[s], y);
    }
    y = fmaf(dsk, uv, y);                       // + D * u_c
    float zv = Z[row * DINNER + d];
    YG[row * DINNER + d] = y * silu_f(zv);      // gate
  }
}

// ---- 6) out_proj GEMM + residual: h = yg(M x 512) @ W^T(512 x 256) + x -----
__global__ __launch_bounds__(128)
void outproj_gemm_kernel(const float* YG, const __bf16* Wb, const float* X,
                         float* HP) {
  const int wave = threadIdx.x >> 5;
  const int m0 = blockIdx.x * 16;
  const int n0 = wave * 64;                     // 4 waves cover N=256
  v8f acc[4] = {};
  for (int k0 = 0; k0 < DINNER; k0 += 32) {
    v16bf a = load_a_f32(YG, m0, k0, DINNER);
#pragma unroll
    for (int j = 0; j < 4; ++j) {
      v16bf b = load_b_bf16(Wb, n0 + j * 16, k0, DINNER);
      acc[j] = __builtin_amdgcn_wmma_f32_16x16x32_bf16(
          false, a, false, b, (short)0, acc[j], false, false);
    }
  }
  const int lane = threadIdx.x & 31;
  const int rbase = m0 + ((lane >> 4) << 3);
#pragma unroll
  for (int j = 0; j < 4; ++j) {
    int col = n0 + j * 16 + (lane & 15);
#pragma unroll
    for (int v = 0; v < 8; ++v) {
      size_t o = (size_t)(rbase + v) * DMODEL + col;
      HP[o] = acc[j][v] + X[o];                 // fused residual
    }
  }
}

// ---- 7) LayerNorm over d_model=256, one wave32 per row ---------------------
__global__ __launch_bounds__(256)
void layernorm_kernel(const float* HP, const float* g, const float* bta,
                      float* OUT) {
  const int wave = threadIdx.x >> 5;
  const int lane = threadIdx.x & 31;
  const int row = blockIdx.x * 8 + wave;
  const float* hp = HP + (size_t)row * DMODEL;
  float v[8], sum = 0.0f, sq = 0.0f;
#pragma unroll
  for (int i = 0; i < 8; ++i) {
    v[i] = hp[lane + i * 32];
    sum += v[i];
    sq = fmaf(v[i], v[i], sq);
  }
#pragma unroll
  for (int m = 16; m >= 1; m >>= 1) {
    sum += __shfl_xor(sum, m, 32);
    sq  += __shfl_xor(sq, m, 32);
  }
  float mu = sum * (1.0f / DMODEL);
  float var = sq * (1.0f / DMODEL) - mu * mu;
  float rstd = rsqrtf(var + 1e-5f);
  float* out = OUT + (size_t)row * DMODEL;
#pragma unroll
  for (int i = 0; i < 8; ++i) {
    int col = lane + i * 32;
    out[col] = (v[i] - mu) * rstd * g[col] + bta[col];
  }
}

// ---------------------------------------------------------------------------
extern "C" void kernel_launch(void* const* d_in, const int* in_sizes, int n_in,
                              void* d_out, int out_size, void* d_ws, size_t ws_size,
                              hipStream_t stream) {
  (void)in_sizes; (void)n_in; (void)out_size; (void)ws_size;
  const float* x     = (const float*)d_in[0];
  const float* w_in  = (const float*)d_in[1];
  const float* cw    = (const float*)d_in[2];
  const float* cb    = (const float*)d_in[3];
  const float* w_xp  = (const float*)d_in[4];
  const float* dtw   = (const float*)d_in[5];
  const float* dtb   = (const float*)d_in[6];
  const float* a_log = (const float*)d_in[7];
  const float* dvec  = (const float*)d_in[8];
  const float* w_out = (const float*)d_in[9];
  const float* ln_g  = (const float*)d_in[10];
  const float* ln_b  = (const float*)d_in[11];
  float* out = (float*)d_out;

  // workspace carve-up (256B aligned); YG reuses the U buffer (U dead after conv)
  char* base = (char*)d_ws;
  size_t off = 0;
  auto carve = [&](size_t bytes) {
    char* p = base + off;
    off = (off + bytes + 255) & ~(size_t)255;
    return p;
  };
  float*  U     = (float*)carve((size_t)MROWS * DINNER * 4);  // later: YG
  float*  Zb    = (float*)carve((size_t)MROWS * DINNER * 4);
  float*  UC    = (float*)carve((size_t)MROWS * DINNER * 4);
  float*  DELTA = (float*)carve((size_t)MROWS * DINNER * 4);
  float*  XD    = (float*)carve((size_t)MROWS * XDBL_N * 4);
  float*  HP    = (float*)carve((size_t)MROWS * DMODEL * 4);
  float*  Pbuf  = (float*)carve((size_t)BATCH * NCH * DSTATE * DINNER * 4);
  float*  Hend  = (float*)carve((size_t)BATCH * NCH * DSTATE * DINNER * 4);
  float*  Hinit = (float*)carve((size_t)BATCH * NCH * DSTATE * DINNER * 4);
  __bf16* WBin  = (__bf16*)carve((size_t)NIN * 2);
  __bf16* WBxp  = (__bf16*)carve((size_t)NXP * 2);
  __bf16* WBout = (__bf16*)carve((size_t)NOUT * 2);
  float*  Aneg  = (float*)carve((size_t)NA * 4);
  float*  YG    = U;

  // 0) weight prep
  prep_kernel<<<1664, 256, 0, stream>>>(w_in, w_xp, w_out, a_log,
                                        WBin, WBxp, WBout, Aneg);
  // 1) in_proj
  inproj_gemm_kernel<<<dim3(MROWS / 16, 4, 1), 128, 0, stream>>>(x, WBin, U, Zb);
  // 2) conv + silu
  conv_silu_kernel<<<(MROWS * DINNER) / 256, 256, 0, stream>>>(U, cw, cb, UC);
  // 3) x_proj
  xproj_gemm_kernel<<<MROWS / (16 * 8), 256, 0, stream>>>(UC, WBxp, XD);
  // 4) dt_proj + softplus
  dtproj_kernel<<<(MROWS * DINNER) / 256, 256, 0, stream>>>(XD, dtw, dtb, DELTA);
  // 5) chunked scan
  scan_pass1_kernel<<<dim3(DINNER / 256, NCH, BATCH), 256, 0, stream>>>(
      DELTA, UC, XD, Aneg, Pbuf, Hend);
  scan_pass2_kernel<<<(BATCH * DINNER) / 256, 256, 0, stream>>>(Pbuf, Hend, Hinit);
  scan_pass3_kernel<<<dim3(DINNER / 256, NCH, BATCH), 256, 0, stream>>>(
      DELTA, UC, XD, Aneg, Hinit, Zb, dvec, YG);
  // 6) out_proj + residual
  outproj_gemm_kernel<<<dim3(MROWS / 16, 1, 1), 128, 0, stream>>>(YG, WBout, x, HP);
  // 7) layernorm
  layernorm_kernel<<<MROWS / 8, 256, 0, stream>>>(HP, ln_g, ln_b, out);
}
